// MultiheadAttention_23742579212817
// MI455X (gfx1250) — matrix operand order repaired and checked by reading, not compiled, not run on hardware
//
#include <hip/hip_runtime.h>

#define EMBED 768
#define HEADS 12
#define HDIM  64
#define BATCH 2
#define SEQ   2048

typedef __attribute__((ext_vector_type(16))) _Float16 v16h;
typedef __attribute__((ext_vector_type(8)))  _Float16 v8h;
typedef __attribute__((ext_vector_type(8)))  float    v8f;
typedef __attribute__((ext_vector_type(4)))  float    v4f;
typedef __attribute__((ext_vector_type(4)))  int      v4i;

#if __has_builtin(__builtin_amdgcn_global_load_async_to_lds_b128) && \
    __has_builtin(__builtin_amdgcn_s_wait_asynccnt)
#define HAVE_ASYNC_LDS 1
#else
#define HAVE_ASYNC_LDS 0
#endif

static __device__ __forceinline__ v16h join16(v8h lo, v8h hi) {
    return __builtin_shufflevector(lo, hi, 0,1,2,3,4,5,6,7,8,9,10,11,12,13,14,15);
}

static __device__ __forceinline__ v8f wmma_f16(v16h a, v16h b, v8f c) {
    // (neg_a, A, neg_b, B, c_mod, C, reuse_a, reuse_b)
    return __builtin_amdgcn_wmma_f32_16x16x32_f16(false, a, false, b, (short)0, c, false, false);
}

static __device__ __forceinline__ v8h cvt8(v4f a, v4f b) {
    v8h r;
    r[0]=(_Float16)a[0]; r[1]=(_Float16)a[1]; r[2]=(_Float16)a[2]; r[3]=(_Float16)a[3];
    r[4]=(_Float16)b[0]; r[5]=(_Float16)b[1]; r[6]=(_Float16)b[2]; r[7]=(_Float16)b[3];
    return r;
}

// ---------------------------------------------------------------------------
// Stage 1: O[b,h,s,d] (f16) = (X @ W^T + bias) * oscale, head-split.
// Block = 256 threads = 8 waves; block tile 64(m) x 64(n); k-step 32.
// ---------------------------------------------------------------------------
#define LDA 48   // halves; 96B row stride (16B multiple)
#define LDB 48

__global__ __launch_bounds__(256)
void qkv_proj_kernel(const float* __restrict__ X,    // [B*S, E]
                     const float* __restrict__ W,    // [E(out), E(in)]
                     const float* __restrict__ bias, // [E]
                     _Float16* __restrict__ O,       // [B,H,S,D]
                     float oscale)
{
    __shared__ __attribute__((aligned(32))) _Float16 lds_a[64 * LDA];
    __shared__ __attribute__((aligned(32))) _Float16 lds_b[64 * LDB];

    const int m0 = blockIdx.x * 64;
    const int n0 = blockIdx.y * 64;

    const int tid  = threadIdx.x;
    const int wave = tid >> 5;
    const int lane = tid & 31;
    const int g    = lane >> 4;   // half-wave group
    const int ln   = lane & 15;

    const int wm = wave & 3;      // 16-row tile within block
    const int wn = wave >> 2;     // 0..1 : owns n columns [wn*32, wn*32+32)

    v8f acc0 = {}; // tile (wm, wn*2)
    v8f acc1 = {}; // tile (wm, wn*2+1)

    const int srow = tid >> 2;        // 0..63
    const int scol = (tid & 3) * 8;   // 0,8,16,24

    for (int k0 = 0; k0 < EMBED; k0 += 32) {
        // stage X[m0+srow][k0+scol..+7] -> f16 LDS
        {
            const float* src = X + (size_t)(m0 + srow) * EMBED + (k0 + scol);
            v4f x0 = *(const v4f*)src;
            v4f x1 = *(const v4f*)(src + 4);
            *(v8h*)&lds_a[srow * LDA + scol] = cvt8(x0, x1);
        }
        // stage W[n0+srow][k0+scol..+7] -> f16 LDS (B[k][n] = W[n][k])
        {
            const float* src = W + (size_t)(n0 + srow) * EMBED + (k0 + scol);
            v4f x0 = *(const v4f*)src;
            v4f x1 = *(const v4f*)(src + 4);
            *(v8h*)&lds_b[srow * LDB + scol] = cvt8(x0, x1);
        }
        __syncthreads();

        // A fragment: lane(g,ln) -> row wm*16+ln, K = {g*16..g*16+15}
        const _Float16* arow = &lds_a[(wm * 16 + ln) * LDA];
        v16h aF = join16(*(const v8h*)&arow[g * 16], *(const v8h*)&arow[g * 16 + 8]);

        // B fragments: lane(g,ln) element i = W[col][k0+g*16+i]
        const _Float16* br0 = &lds_b[(wn * 32 + ln) * LDB];
        const _Float16* br1 = &lds_b[(wn * 32 + 16 + ln) * LDB];
        v16h bF0 = join16(*(const v8h*)&br0[g * 16], *(const v8h*)&br0[g * 16 + 8]);
        v16h bF1 = join16(*(const v8h*)&br1[g * 16], *(const v8h*)&br1[g * 16 + 8]);

        acc0 = wmma_f16(aF, bF0, acc0);
        acc1 = wmma_f16(aF, bF1, acc1);
        __syncthreads();
    }

    // epilogue: +bias, *oscale, store f16 [B,H,S,D]
    const int head = n0 / HDIM;
    #pragma unroll
    for (int t = 0; t < 2; ++t) {
        v8f acc = t ? acc1 : acc0;
        const int col  = (wn * 2 + t) * 16 + ln;          // 0..63 inside head
        const float bv = bias[n0 + col];
        #pragma unroll
        for (int j = 0; j < 8; ++j) {
            const int m  = m0 + wm * 16 + g * 8 + j;
            const int bb = m >> 11;          // /SEQ
            const int s  = m & (SEQ - 1);
            const float v = (acc[j] + bv) * oscale;
            O[(((size_t)bb * HEADS + head) * SEQ + s) * HDIM + col] = (_Float16)v;
        }
    }
}

// ---------------------------------------------------------------------------
// Stage 2: flash attention. Block = (b, h, 128 q-rows), 8 waves.
// Wave w owns q-tile w (16 rows) and the FULL d=64 width (4 accumulators):
// no redundant score/softmax work. Key tiles of 32.
// ---------------------------------------------------------------------------
#define LDQ 80
#define LDK 80
#define LDV 48
#define LDP 32

__global__ __launch_bounds__(256)
void attention_kernel(const _Float16* __restrict__ Qh,   // [B,H,S,D] (pre-scaled 1/8)
                      const _Float16* __restrict__ Kh,   // [B,H,S,D]
                      const _Float16* __restrict__ Vh,   // [B,H,S,D]
                      float* __restrict__ Out)           // [B,S,E]
{
    __shared__ __attribute__((aligned(32))) _Float16 lds_q[128 * LDQ];
    __shared__ __attribute__((aligned(32))) _Float16 lds_k[32 * LDK];
    __shared__ __attribute__((aligned(32))) _Float16 lds_vt[64 * LDV];
    __shared__ __attribute__((aligned(32))) _Float16 lds_p[8][16 * LDP];

    const int bh = blockIdx.y;
    const int b  = bh / HEADS;
    const int h  = bh % HEADS;
    const int q0 = blockIdx.x * 128;

    const size_t base = (size_t)bh * SEQ * HDIM;
    const _Float16* Qp = Qh + base;
    const _Float16* Kp = Kh + base;
    const _Float16* Vp = Vh + base;

    const int tid  = threadIdx.x;
    const int wave = tid >> 5;
    const int lane = tid & 31;
    const int g    = lane >> 4;
    const int ln   = lane & 15;

    // stage Q (128x64 halves)
    {
        const int r = tid >> 1;
        const int c = (tid & 1) * 32;
        const v8h* src = (const v8h*)(Qp + (size_t)(q0 + r) * HDIM + c);
        v8h* dst = (v8h*)&lds_q[r * LDQ + c];
        dst[0] = src[0];
        dst[1] = src[1];
        dst[2] = src[2];
        dst[3] = src[3];
    }
    __syncthreads();

    // Q A-fragments for d-chunks 0 and 1 (kept in registers whole kernel)
    v16h aQ0, aQ1;
    {
        const _Float16* qr = &lds_q[(wave * 16 + ln) * LDQ];
        aQ0 = join16(*(const v8h*)&qr[g * 16],      *(const v8h*)&qr[g * 16 + 8]);
        aQ1 = join16(*(const v8h*)&qr[32 + g * 16], *(const v8h*)&qr[32 + g * 16 + 8]);
    }

    v8f oacc[4] = {};                 // d-tiles 0..3 (16 cols each)
    float mrow[8], lrow[8];
    #pragma unroll
    for (int j = 0; j < 8; ++j) { mrow[j] = -1e30f; lrow[j] = 0.0f; }

    const int kr = tid >> 3;        // 0..31
    const int kc = (tid & 7) * 8;   // 0..56

    for (int kt = 0; kt < SEQ; kt += 32) {
        // ---- stage K [kk][d] (async direct-to-LDS when available) and
        //      V transposed [d][kk] (manual, needs element transpose)
        {
            const size_t off = (size_t)(kt + kr) * HDIM + kc;
#if HAVE_ASYNC_LDS
            __builtin_amdgcn_global_load_async_to_lds_b128(
                (v4i*)(Kp + off),
                (v4i*)&lds_k[kr * LDK + kc],
                0, 0);
#else
            *(v8h*)&lds_k[kr * LDK + kc] = *(const v8h*)(Kp + off);
#endif
            v8h vv = *(const v8h*)(Vp + off);
            #pragma unroll
            for (int i = 0; i < 8; ++i)
                lds_vt[(kc + i) * LDV + kr] = vv[i];
            if (kt + 32 < SEQ) {
                __builtin_prefetch(Kp + off + 32 * HDIM, 0, 3);
                __builtin_prefetch(Vp + off + 32 * HDIM, 0, 3);
            }
        }
#if HAVE_ASYNC_LDS
        __builtin_amdgcn_s_wait_asynccnt(0);
#endif
        __syncthreads();

        // ---- scores: two 16x16 tiles over 32 keys, accumulate over d=64
        v8f s0 = {}, s1 = {};
        {
            const _Float16* k0r = &lds_k[ln * LDK];
            const _Float16* k1r = &lds_k[(16 + ln) * LDK];
            v16h b00 = join16(*(const v8h*)&k0r[g * 16],      *(const v8h*)&k0r[g * 16 + 8]);
            v16h b01 = join16(*(const v8h*)&k0r[32 + g * 16], *(const v8h*)&k0r[32 + g * 16 + 8]);
            v16h b10 = join16(*(const v8h*)&k1r[g * 16],      *(const v8h*)&k1r[g * 16 + 8]);
            v16h b11 = join16(*(const v8h*)&k1r[32 + g * 16], *(const v8h*)&k1r[32 + g * 16 + 8]);
            s0 = wmma_f16(aQ0, b00, s0);
            s0 = wmma_f16(aQ1, b01, s0);
            s1 = wmma_f16(aQ0, b10, s1);
            s1 = wmma_f16(aQ1, b11, s1);
        }

        // ---- online softmax; row r = g*8+j lives in half-wave g across 16 lanes
        _Float16* pw = &lds_p[wave][0];
        #pragma unroll
        for (int j = 0; j < 8; ++j) {
            float mj = fmaxf(s0[j], s1[j]);
            mj = fmaxf(mj, __shfl_xor(mj, 1, 32));
            mj = fmaxf(mj, __shfl_xor(mj, 2, 32));
            mj = fmaxf(mj, __shfl_xor(mj, 4, 32));
            mj = fmaxf(mj, __shfl_xor(mj, 8, 32));
            const float mnew  = fmaxf(mrow[j], mj);
            const float alpha = __expf(mrow[j] - mnew);
            const float p0 = __expf(s0[j] - mnew);
            const float p1 = __expf(s1[j] - mnew);
            float rs = p0 + p1;
            rs += __shfl_xor(rs, 1, 32);
            rs += __shfl_xor(rs, 2, 32);
            rs += __shfl_xor(rs, 4, 32);
            rs += __shfl_xor(rs, 8, 32);
            lrow[j] = lrow[j] * alpha + rs;
            mrow[j] = mnew;
            #pragma unroll
            for (int dt = 0; dt < 4; ++dt) oacc[dt][j] *= alpha;
            pw[(g * 8 + j) * LDP + ln]      = (_Float16)p0;   // D-layout -> LDS
            pw[(g * 8 + j) * LDP + 16 + ln] = (_Float16)p1;
        }

        // ---- re-read P in A-fragment layout (wave-private region, no barrier)
        const _Float16* pr = &lds_p[wave][ln * LDP];
        v16h pA = join16(*(const v8h*)&pr[g * 16], *(const v8h*)&pr[g * 16 + 8]);

        // ---- P @ V for all four d-column tiles
        #pragma unroll
        for (int dt = 0; dt < 4; ++dt) {
            const _Float16* vr = &lds_vt[(dt * 16 + ln) * LDV];
            v16h bv = join16(*(const v8h*)&vr[g * 16], *(const v8h*)&vr[g * 16 + 8]);
            oacc[dt] = wmma_f16(pA, bv, oacc[dt]);
        }
        __syncthreads();
    }

    // ---- normalize and store fp32 [B,S,E]
    #pragma unroll
    for (int j = 0; j < 8; ++j) {
        const float inv = 1.0f / lrow[j];
        const int s = q0 + wave * 16 + g * 8 + j;
        float* orow = Out + ((size_t)b * SEQ + s) * EMBED + h * HDIM;
        #pragma unroll
        for (int dt = 0; dt < 4; ++dt)
            orow[dt * 16 + ln] = oacc[dt][j] * inv;
    }
}

// ---------------------------------------------------------------------------
extern "C" void kernel_launch(void* const* d_in, const int* in_sizes, int n_in,
                              void* d_out, int out_size, void* d_ws, size_t ws_size,
                              hipStream_t stream) {
    const float* query = (const float*)d_in[0];
    const float* key   = (const float*)d_in[1];
    const float* value = (const float*)d_in[2];
    const float* Wq = (const float*)d_in[3];
    const float* bq = (const float*)d_in[4];
    const float* bk = nullptr; (void)bk;
    const float* Wk = (const float*)d_in[5];
    const float* bk2 = (const float*)d_in[6];
    const float* Wv = (const float*)d_in[7];
    const float* bv = (const float*)d_in[8];

    const size_t per = (size_t)BATCH * HEADS * SEQ * HDIM; // elements
    _Float16* Qh = (_Float16*)d_ws;
    _Float16* Kh = Qh + per;
    _Float16* Vh = Kh + per;

    dim3 g1(BATCH * SEQ / 64, EMBED / 64); // (64, 12)
    // fold softmax 1/sqrt(D)=0.125 into Q
    qkv_proj_kernel<<<g1, 256, 0, stream>>>(query, Wq, bq, Qh, 0.125f);
    qkv_proj_kernel<<<g1, 256, 0, stream>>>(key,   Wk, bk2, Kh, 1.0f);
    qkv_proj_kernel<<<g1, 256, 0, stream>>>(value, Wv, bv, Vh, 1.0f);

    dim3 g2(SEQ / 128, BATCH * HEADS);     // (16, 24)
    attention_kernel<<<g2, 256, 0, stream>>>(Qh, Kh, Vh, (float*)d_out);
}
